// PTB2bBmm_7224134992441
// MI455X (gfx1250) — matrix operand order repaired and checked by reading, not compiled, not run on hardware
//
#include <hip/hip_runtime.h>

#define S_LEN 2048
#define D_HEAD 128
#define BH 24
#define ALPHA0 0.08838834764831845f   // 1/sqrt(128)

typedef __attribute__((ext_vector_type(16))) __bf16 bf16x16;
typedef __attribute__((ext_vector_type(8)))  float  f32x8;

union Frag {
  bf16x16 v;
  uint4 q[2];
  unsigned short u[16];
};

__device__ __forceinline__ unsigned short f2b(float f) {
  __bf16 h = (__bf16)f;
  return __builtin_bit_cast(unsigned short, h);
}

// ---- LDS layout (dynamic shared memory, base offset 0) ----
// strides in elements; padded for bank-conflict-free access, 16B aligned.
#define KSTR 136                 // K tile row stride (u16): 128 + 8
#define VSTR 40                  // V^T tile row stride (u16): 32 + 8
#define PSTR 40                  // P scratch row stride (u16): 32 + 8
#define BSTR 36                  // bias row stride (f32): 32 + 4

#define KT_OFF   0               // 32 * 136 * 2  = 8704 B
#define VT_OFF   8704            // 128 * 40 * 2  = 10240 B
#define PW_OFF   18944           // 8 * 16 * 40*2 = 10240 B
#define BIAS_OFF 29184           // 8 waves * 2 bufs * 16 * 36 * 4 = 36864 B
#define LDS_TOTAL 66048

// Async copy of one 16B packet from global to LDS (ASYNCcnt-tracked).
__device__ __forceinline__ void async_b128_to_lds(unsigned lds_byte_off,
                                                  unsigned gbyte_off,
                                                  const float* sbase) {
  asm volatile("global_load_async_to_lds_b128 %0, %1, %2 th:TH_LOAD_NT"
               :: "v"(lds_byte_off), "v"(gbyte_off), "s"(sbase)
               : "memory");
}

__global__ __launch_bounds__(256)
void sigmoid_attn_wmma(const float* __restrict__ qg,
                       const float* __restrict__ kg,
                       const float* __restrict__ vg,
                       const float* __restrict__ bg,
                       float* __restrict__ og) {
  extern __shared__ __align__(16) unsigned char smem[];
  unsigned short* Kt = (unsigned short*)(smem + KT_OFF);  // bf16 [key][d]
  unsigned short* Vt = (unsigned short*)(smem + VT_OFF);  // bf16 [d][key]
  unsigned short* Pw = (unsigned short*)(smem + PW_OFF);  // per-wave P scratch
  float*          Bb = (float*)(smem + BIAS_OFF);         // per-wave bias dbuf

  const int tid  = threadIdx.x;
  const int lane = tid & 31;
  const int wave = tid >> 5;
  const int lhi  = lane >> 4;   // half-wave: 0 or 1
  const int llo  = lane & 15;

  const int bh     = blockIdx.x >> 4;   // 0..23  (b*H + h)
  const int rowblk = blockIdx.x & 15;   // 0..15
  const int qbase  = rowblk * 128 + wave * 16;

  const float* qb = qg + (size_t)bh * S_LEN * D_HEAD;
  const float* kb = kg + (size_t)bh * S_LEN * D_HEAD;
  const float* vb = vg + (size_t)bh * S_LEN * D_HEAD;
  const float* bb = bg + (size_t)bh * S_LEN * S_LEN;
  float*       ob = og + (size_t)bh * S_LEN * D_HEAD;

  // Per-lane pieces of the bias async-copy pattern (4 instrs cover 16x32 f32)
  const unsigned pr_sub = lane >> 3;        // row sub-index 0..3
  const unsigned pr_p   = (lane & 7) * 16;  // 16B packet within 128B row

  const float* bias_row0 = bb + (size_t)qbase * S_LEN;  // + t0 added per chunk

  // ---- Load Q as 4 bf16 A-fragments (16 rows x 32 K-slice each) ----
  Frag qa[4];
  {
    const float* qp = qb + (size_t)(qbase + llo) * D_HEAD;
    #pragma unroll
    for (int kc = 0; kc < 4; ++kc) {
      #pragma unroll
      for (int r = 0; r < 2; ++r) {
        const int d0 = kc * 32 + r * 16 + lhi * 8;
        float4 f0 = *(const float4*)(qp + d0);
        float4 f1 = *(const float4*)(qp + d0 + 4);
        const int e = r * 8;
        qa[kc].u[e+0] = f2b(f0.x); qa[kc].u[e+1] = f2b(f0.y);
        qa[kc].u[e+2] = f2b(f0.z); qa[kc].u[e+3] = f2b(f0.w);
        qa[kc].u[e+4] = f2b(f1.x); qa[kc].u[e+5] = f2b(f1.y);
        qa[kc].u[e+6] = f2b(f1.z); qa[kc].u[e+7] = f2b(f1.w);
      }
    }
  }

  // O accumulator: 8 column tiles of 16x16 f32 (C/D layout), 64 VGPRs.
  f32x8 o[8];
  #pragma unroll
  for (int c = 0; c < 8; ++c) o[c] = (f32x8){0,0,0,0,0,0,0,0};

  const int skey = tid >> 3;         // staging: key 0..31
  const int sd0  = (tid & 7) * 16;   // staging: d offset 0..112

  // ---- prefetch bias tile for chunk 0 into buffer 0 ----
  #pragma unroll
  for (unsigned i = 0; i < 4; ++i) {
    const unsigned r = i * 4 + pr_sub;                  // 0..15
    async_b128_to_lds(BIAS_OFF + (unsigned)(wave * 2 + 0) * 2304u
                          + r * 144u + pr_p,
                      r * (S_LEN * 4u) + pr_p, bias_row0);
  }

  for (int t0 = 0; t0 < S_LEN; t0 += 32) {
    const int buf = (t0 >> 5) & 1;
    // ---- prefetch next chunk's bias into the other buffer ----
    if (t0 + 32 < S_LEN) {
      #pragma unroll
      for (unsigned i = 0; i < 4; ++i) {
        const unsigned r = i * 4 + pr_sub;
        async_b128_to_lds(BIAS_OFF + (unsigned)(wave * 2 + (buf ^ 1)) * 2304u
                              + r * 144u + pr_p,
                          r * (S_LEN * 4u) + pr_p,
                          bias_row0 + (t0 + 32));
      }
    }

    __syncthreads();  // previous iteration's K/V fragment reads done
    // ---- cooperatively stage K chunk (bf16, row-major) ----
    {
      const float* kp = kb + (size_t)(t0 + skey) * D_HEAD + sd0;
      float4 f0 = *(const float4*)(kp + 0);
      float4 f1 = *(const float4*)(kp + 4);
      float4 f2 = *(const float4*)(kp + 8);
      float4 f3 = *(const float4*)(kp + 12);
      unsigned short tmp[16] = {
        f2b(f0.x), f2b(f0.y), f2b(f0.z), f2b(f0.w),
        f2b(f1.x), f2b(f1.y), f2b(f1.z), f2b(f1.w),
        f2b(f2.x), f2b(f2.y), f2b(f2.z), f2b(f2.w),
        f2b(f3.x), f2b(f3.y), f2b(f3.z), f2b(f3.w)};
      uint4* dst = (uint4*)&Kt[skey * KSTR + sd0];
      dst[0] = *(const uint4*)&tmp[0];
      dst[1] = *(const uint4*)&tmp[8];
    }
    // ---- stage V chunk transposed: Vt[d][key] (bf16) ----
    {
      const float* vp = vb + (size_t)(t0 + skey) * D_HEAD + sd0;
      float4 f0 = *(const float4*)(vp + 0);
      float4 f1 = *(const float4*)(vp + 4);
      float4 f2 = *(const float4*)(vp + 8);
      float4 f3 = *(const float4*)(vp + 12);
      float fv[16] = {f0.x,f0.y,f0.z,f0.w, f1.x,f1.y,f1.z,f1.w,
                      f2.x,f2.y,f2.z,f2.w, f3.x,f3.y,f3.z,f3.w};
      #pragma unroll
      for (int i = 0; i < 16; ++i)
        Vt[(sd0 + i) * VSTR + skey] = f2b(fv[i]);
    }
    __syncthreads();

    // ---- wait for this chunk's bias copies (next chunk's may stay in flight)
    if (t0 + 32 < S_LEN)
      asm volatile("s_wait_asynccnt 0x4" ::: "memory");
    else
      asm volatile("s_wait_asynccnt 0x0" ::: "memory");

    const float* bias_tile = Bb + (wave * 2 + buf) * (16 * BSTR);

    // ---- scores for two 16-key tiles, fused bias + sigmoid ----
    #pragma unroll
    for (int j = 0; j < 2; ++j) {
      f32x8 sc = (f32x8){0,0,0,0,0,0,0,0};
      #pragma unroll
      for (int kc = 0; kc < 4; ++kc) {
        Frag bk;
        const uint4* src =
            (const uint4*)&Kt[(j * 16 + llo) * KSTR + kc * 32 + lhi * 8];
        bk.q[0] = src[0];
        bk.q[1] = src[2];
        sc = __builtin_amdgcn_wmma_f32_16x16x32_bf16(
            false, qa[kc].v, false, bk.v, (short)0, sc, false, false);
      }
      #pragma unroll
      for (int v = 0; v < 8; ++v) {
        const int m = v + lhi * 8;  // C/D layout: row = vgpr + 8*half
        float bval = bias_tile[m * BSTR + j * 16 + llo];
        float x = ALPHA0 * sc[v] + bval;
        // sigmoid via hardware exp + rcp (no IEEE divide expansion)
        float p = __builtin_amdgcn_rcpf(1.0f + __expf(-x));
        Pw[wave * (16 * PSTR) + m * PSTR + j * 16 + llo] = f2b(p);
      }
    }
    // P scratch is wave-private; DS ops are wave-ordered, drain DScnt so
    // every lane sees every other lane's stores.
    asm volatile("s_wait_dscnt 0" ::: "memory");

    // ---- reload P in A-fragment layout (cross-lane transpose via LDS) ----
    Frag pa;
    {
      const unsigned short* pwb = Pw + wave * (16 * PSTR);
      const uint4* r0 = (const uint4*)&pwb[llo * PSTR + lhi * 8];
      const uint4* r1 = (const uint4*)&pwb[llo * PSTR + 16 + lhi * 8];
      pa.q[0] = r0[0];
      pa.q[1] = r1[0];
    }
    // ---- O += P(16x32) @ V(32x16) for 8 d-column tiles ----
    #pragma unroll
    for (int c = 0; c < 8; ++c) {
      Frag vf;
      const uint4* src =
          (const uint4*)&Vt[(c * 16 + llo) * VSTR + lhi * 8];
      vf.q[0] = src[0];
      vf.q[1] = src[2];
      o[c] = __builtin_amdgcn_wmma_f32_16x16x32_bf16(
          false, pa.v, false, vf.v, (short)0, o[c], false, false);
    }
  }

  // ---- store O (nontemporal, coalesced per half-wave row) ----
  #pragma unroll
  for (int c = 0; c < 8; ++c) {
    #pragma unroll
    for (int v = 0; v < 8; ++v) {
      const int row = qbase + v + lhi * 8;
      __builtin_nontemporal_store(o[c][v],
                                  ob + (size_t)row * D_HEAD + c * 16 + llo);
    }
  }
}

extern "C" void kernel_launch(void* const* d_in, const int* in_sizes, int n_in,
                              void* d_out, int out_size, void* d_ws, size_t ws_size,
                              hipStream_t stream) {
  (void)in_sizes; (void)n_in; (void)out_size; (void)d_ws; (void)ws_size;
  const float* q = (const float*)d_in[0];
  const float* k = (const float*)d_in[1];
  const float* v = (const float*)d_in[2];
  const float* b = (const float*)d_in[3];
  float* out = (float*)d_out;

  dim3 grid(BH * (S_LEN / 128));  // 24 * 16 = 384 workgroups
  dim3 block(256);                // 8 waves (wave32)
  hipLaunchKernelGGL(sigmoid_attn_wmma, grid, block, LDS_TOTAL, stream,
                     q, k, v, b, out);
}
